// TemporalDecoder_34316788695759
// MI455X (gfx1250) — compile-verified, matrix operand order checked by reading, not run on hardware
//
#include <hip/hip_runtime.h>

typedef unsigned short u16;
typedef unsigned int u32;

#define B_TOT   2048
#define LATENT  128
#define HID     512
#define GATES   1536
#define T_STEPS 128
#define NUMF    128
#define MT      16      // batch rows per workgroup
#define SF      516     // f32 LDS row stride (padded)
#define SB      528     // bf16 LDS row stride (padded, 16B-multiple)

typedef __attribute__((ext_vector_type(16))) __bf16 v16bf;
typedef __attribute__((ext_vector_type(8)))  float  v8f;

__device__ __forceinline__ u16 f2bf(float f) {
    u32 u = __float_as_uint(f);
    u += 0x7fffu + ((u >> 16) & 1u);            // round-to-nearest-even
    return (u16)(u >> 16);
}
__device__ __forceinline__ float bf2f(u16 h) {
    return __uint_as_float(((u32)h) << 16);
}
__device__ __forceinline__ float sigm(float x)  { return 1.0f / (1.0f + __expf(-x)); }
__device__ __forceinline__ float lrelu(float x) { return x > 0.0f ? x : 0.2f * x; }

// 16-bit A/B fragment: per lane two contiguous 8-element (16B) runs:
//   elems 0..7  = K (k0 + 8*half) .. +7
//   elems 8..15 = K (k0 + 16 + 8*half) .. +7
__device__ __forceinline__ v16bf ldfrag(const u16* rowbase, int k0, int half) {
    const u16* q = rowbase + k0 + (half << 3);
    union { uint4 u[2]; v16bf v; } un;
    un.u[0] = *(const uint4*)(q);
    un.u[1] = *(const uint4*)(q + 16);
    return un.v;
}

__device__ __forceinline__ v8f wmma_bf16(v16bf a, v16bf b, v8f c) {
    return __builtin_amdgcn_wmma_f32_16x16x32_bf16(false, a, false, b, (short)0, c,
                                                   false, false);
}

// ---------------- precompute kernels ----------------

__global__ void k_cvt_bf16(const float* __restrict__ src, u16* __restrict__ dst, int n) {
    int i = blockIdx.x * blockDim.x + threadIdx.x;
    if (i < n) dst[i] = f2bf(src[i]);
}

// hid[b][j] = leaky_relu(e[b,:] . fc1_w[j,:] + fc1_b[j])
__global__ void k_fc1(const float* __restrict__ e, const float* __restrict__ w,
                      const float* __restrict__ b, float* __restrict__ hid) {
    int o = blockIdx.x * blockDim.x + threadIdx.x;     // 2048*512
    int bb = o >> 9, j = o & 511;
    const float* er = e + (size_t)bb * LATENT;
    const float* wr = w + (size_t)j * LATENT;
    float s = b[j];
    #pragma unroll 4
    for (int k = 0; k < LATENT; ++k) s += er[k] * wr[k];
    hid[o] = lrelu(s);
}

// h0_flat[b][j] = hid[b,:] . fc2_w[j,:] + fc2_b[j]   (j in [0,1024))
__global__ void k_fc2(const float* __restrict__ hid, const float* __restrict__ w,
                      const float* __restrict__ b, float* __restrict__ h0) {
    int o = blockIdx.x * blockDim.x + threadIdx.x;     // 2048*1024
    int bb = o >> 10, j = o & 1023;
    const float* hr = hid + (size_t)bb * HID;
    const float* wr = w + (size_t)j * HID;
    float s = b[j];
    #pragma unroll 4
    for (int k = 0; k < HID; ++k) s += hr[k] * wr[k];
    h0[o] = s;
}

// gi0[b][g] = e[b,:] . w_ih0[g,:] + b_ih0[g]   (loop-invariant across the scan)
__global__ void k_gi0(const float* __restrict__ e, const float* __restrict__ w,
                      const float* __restrict__ b, float* __restrict__ gi0) {
    int o = blockIdx.x * blockDim.x + threadIdx.x;     // 2048*1536
    int bb = o / GATES, j = o - bb * GATES;
    const float* er = e + (size_t)bb * LATENT;
    const float* wr = w + (size_t)j * LATENT;
    float s = b[j];
    #pragma unroll 4
    for (int k = 0; k < LATENT; ++k) s += er[k] * wr[k];
    gi0[o] = s;
}

// ---------------- main persistent scan kernel ----------------

__global__ __launch_bounds__(256, 1)
void td_scan(const float* __restrict__ gi0, const float* __restrict__ h0ws,
             const u16* __restrict__ whh0, const u16* __restrict__ wih1,
             const u16* __restrict__ whh1, const u16* __restrict__ wvw,
             const u16* __restrict__ wt1w, const u16* __restrict__ wmw,
             const float* __restrict__ bhh0, const float* __restrict__ bih1,
             const float* __restrict__ bhh1, const float* __restrict__ bv,
             const float* __restrict__ bt1, const float* __restrict__ wt2,
             const float* __restrict__ bt2, const float* __restrict__ bm,
             float* __restrict__ out_tn, float* __restrict__ out_u,
             float* __restrict__ out_m, int* __restrict__ out_seq) {
    extern __shared__ char smem[];
    float* s_h0f = (float*)smem;                 // [MT][SF] layer-0 state f32
    float* s_h1f = s_h0f + MT * SF;              // [MT][SF] layer-1 state f32
    u16*   s_h0b = (u16*)(s_h1f + MT * SF);      // [MT][SB] bf16 mirror (A operand)
    u16*   s_h1b = s_h0b + MT * SB;
    u16*   s_x1b = s_h1b + MT * SB;              // fresh h1 (GRU1 input)
    u16*   s_h2b = s_x1b + MT * SB;              // fresh h2 (head input)
    float* s_tn  = (float*)(s_h2b + MT * SB);    // [MT][128]
    float* s_m   = s_tn + MT * NUMF;             // [MT][128]
    float* s_t1  = s_m + MT * NUMF;              // [MT][64]
    int*   s_act = (int*)(s_t1 + MT * 64);       // [MT]
    int*   s_len = s_act + MT;                   // [MT]
    int*   s_stop= s_len + MT;                   // [MT]

    const int tid  = threadIdx.x;
    const int wave = tid >> 5;
    const int lane = tid & 31;
    const int half = lane >> 4;
    const int nlan = lane & 15;
    const int b0   = blockIdx.x * MT;

    // ---- init: load h0 (reshape(2,B,H) of the [B,2H] fc output = flat view) ----
    for (int idx = tid; idx < MT * HID; idx += 256) {
        int r = idx >> 9, c = idx & 511;
        float v0 = h0ws[(size_t)0 * B_TOT * HID + (size_t)(b0 + r) * HID + c];
        float v1 = h0ws[(size_t)1 * B_TOT * HID + (size_t)(b0 + r) * HID + c];
        s_h0f[r * SF + c] = v0;  s_h0b[r * SB + c] = f2bf(v0);
        s_h1f[r * SF + c] = v1;  s_h1b[r * SB + c] = f2bf(v1);
    }
    if (tid < MT) { s_act[tid] = 1; s_len[tid] = 0; s_stop[tid] = 0; }
    __syncthreads();

    #pragma unroll 1
    for (int t = 0; t < T_STEPS; ++t) {
        // ===== Phase 1: GRU layer 0 -> fresh h1 into s_x1b =====
        #pragma unroll 1
        for (int jt = wave; jt < HID / 16; jt += 8) {
            const int n0 = jt * 16;
            v8f ar = {0.f,0.f,0.f,0.f,0.f,0.f,0.f,0.f};
            v8f az = ar, an = ar;
            const u16* arow = s_h0b + (size_t)nlan * SB;
            const u16* br_r = whh0 + (size_t)(       n0 + nlan) * HID;
            const u16* br_z = whh0 + (size_t)( 512 + n0 + nlan) * HID;
            const u16* br_n = whh0 + (size_t)(1024 + n0 + nlan) * HID;
            for (int k0 = 0; k0 < HID; k0 += 32) {
                v16bf a = ldfrag(arow, k0, half);
                ar = wmma_bf16(a, ldfrag(br_r, k0, half), ar);
                az = wmma_bf16(a, ldfrag(br_z, k0, half), az);
                an = wmma_bf16(a, ldfrag(br_n, k0, half), an);
            }
            const int cn = n0 + nlan;
            const float bR = bhh0[cn], bZ = bhh0[512 + cn], bN = bhh0[1024 + cn];
            #pragma unroll
            for (int i = 0; i < 8; ++i) {
                const int mr = i + 8 * half;
                const size_t g = (size_t)(b0 + mr) * GATES;
                float r  = sigm(gi0[g + cn]        + ar[i] + bR);
                float z  = sigm(gi0[g + 512 + cn]  + az[i] + bZ);
                float nn = tanhf(gi0[g + 1024 + cn] + r * (an[i] + bN));
                float hp = s_h0f[mr * SF + cn];
                float h1 = (1.0f - z) * nn + z * hp;
                s_x1b[mr * SB + cn] = f2bf(h1);
            }
        }
        __syncthreads();

        // ===== Phase 2: GRU layer 1 -> fresh h2 into s_h2b =====
        #pragma unroll 1
        for (int jt = wave; jt < HID / 16; jt += 8) {
            const int n0 = jt * 16;
            v8f xr = {0.f,0.f,0.f,0.f,0.f,0.f,0.f,0.f};
            v8f xz = xr, xn = xr, hr = xr, hz = xr, hn = xr;
            const u16* axr = s_x1b + (size_t)nlan * SB;
            const u16* ahr = s_h1b + (size_t)nlan * SB;
            const u16* bi_r = wih1 + (size_t)(       n0 + nlan) * HID;
            const u16* bi_z = wih1 + (size_t)( 512 + n0 + nlan) * HID;
            const u16* bi_n = wih1 + (size_t)(1024 + n0 + nlan) * HID;
            const u16* bh_r = whh1 + (size_t)(       n0 + nlan) * HID;
            const u16* bh_z = whh1 + (size_t)( 512 + n0 + nlan) * HID;
            const u16* bh_n = whh1 + (size_t)(1024 + n0 + nlan) * HID;
            for (int k0 = 0; k0 < HID; k0 += 32) {
                v16bf ax = ldfrag(axr, k0, half);
                v16bf ah = ldfrag(ahr, k0, half);
                xr = wmma_bf16(ax, ldfrag(bi_r, k0, half), xr);
                xz = wmma_bf16(ax, ldfrag(bi_z, k0, half), xz);
                xn = wmma_bf16(ax, ldfrag(bi_n, k0, half), xn);
                hr = wmma_bf16(ah, ldfrag(bh_r, k0, half), hr);
                hz = wmma_bf16(ah, ldfrag(bh_z, k0, half), hz);
                hn = wmma_bf16(ah, ldfrag(bh_n, k0, half), hn);
            }
            const int cn = n0 + nlan;
            const float biR = bih1[cn], biZ = bih1[512 + cn], biN = bih1[1024 + cn];
            const float bhR = bhh1[cn], bhZ = bhh1[512 + cn], bhN = bhh1[1024 + cn];
            #pragma unroll
            for (int i = 0; i < 8; ++i) {
                const int mr = i + 8 * half;
                float r  = sigm(xr[i] + biR + hr[i] + bhR);
                float z  = sigm(xz[i] + biZ + hz[i] + bhZ);
                float nn = tanhf(xn[i] + biN + r * (hn[i] + bhN));
                float hp = s_h1f[mr * SF + cn];
                float h2 = (1.0f - z) * nn + z * hp;
                s_h2b[mr * SB + cn] = f2bf(h2);
            }
        }
        __syncthreads();

        // ===== Phase 3: heads from fresh h2 (20 output tiles) =====
        #pragma unroll 1
        for (int tt = wave; tt < 20; tt += 8) {
            v8f acc = {0.f,0.f,0.f,0.f,0.f,0.f,0.f,0.f};
            const u16* arow = s_h2b + (size_t)nlan * SB;
            const u16* wrow;
            int n0l;
            if (tt < 8)       { n0l = tt * 16;        wrow = wvw  + (size_t)(n0l + nlan) * HID; }
            else if (tt < 16) { n0l = (tt - 8) * 16;  wrow = wmw  + (size_t)(n0l + nlan) * HID; }
            else              { n0l = (tt - 16) * 16; wrow = wt1w + (size_t)(n0l + nlan) * HID; }
            for (int k0 = 0; k0 < HID; k0 += 32)
                acc = wmma_bf16(ldfrag(arow, k0, half), ldfrag(wrow, k0, half), acc);
            const int c = n0l + nlan;
            if (tt < 8) {
                const float b = bv[c];
                #pragma unroll
                for (int i = 0; i < 8; ++i) s_tn[(i + 8 * half) * NUMF + c] = sigm(acc[i] + b);
            } else if (tt < 16) {
                const float b = bm[c];
                #pragma unroll
                for (int i = 0; i < 8; ++i) s_m[(i + 8 * half) * NUMF + c] = acc[i] + b;
            } else {
                const float b = bt1[c];
                #pragma unroll
                for (int i = 0; i < 8; ++i) s_t1[(i + 8 * half) * 64 + c] = lrelu(acc[i] + b);
            }
        }
        __syncthreads();

        // ===== State commit (masked), u head, stop flag, global writes =====
        for (int idx = tid; idx < MT * HID; idx += 256) {
            int r = idx >> 9, c = idx & 511;
            int act = s_act[r];
            float h1 = bf2f(s_x1b[r * SB + c]);
            float h2 = bf2f(s_h2b[r * SB + c]);
            float n0v = act ? h1 : s_h0f[r * SF + c];
            float n1v = act ? h2 : s_h1f[r * SF + c];
            s_h0f[r * SF + c] = n0v;  s_h0b[r * SB + c] = f2bf(n0v);
            s_h1f[r * SF + c] = n1v;  s_h1b[r * SB + c] = f2bf(n1v);
        }
        if (tid < MT) {
            float s = bt2[0];
            #pragma unroll 4
            for (int k = 0; k < 64; ++k) s += s_t1[tid * 64 + k] * wt2[k];
            out_u[(size_t)(b0 + tid) * T_STEPS + t] = sigm(s);
            int stop = 1;
            for (int k = 0; k < NUMF; ++k) stop &= (s_m[tid * NUMF + k] < 0.0f) ? 1 : 0;
            s_stop[tid] = stop;
        }
        for (int idx = tid; idx < MT * NUMF; idx += 256) {
            int rr = idx >> 7, cc = idx & 127;
            size_t o = (size_t)(b0 + rr) * T_STEPS * NUMF + (size_t)t * NUMF + cc;
            out_tn[o] = s_tn[idx];
            out_m[o]  = s_m[idx];
        }
        __syncthreads();
        if (tid < MT) {
            if (s_act[tid]) {
                s_len[tid] += 1;
                if (s_stop[tid]) s_act[tid] = 0;
            }
        }
        __syncthreads();
    }

    if (tid < MT) out_seq[b0 + tid] = s_len[tid];
}

// ---------------- host launcher ----------------

extern "C" void kernel_launch(void* const* d_in, const int* in_sizes, int n_in,
                              void* d_out, int out_size, void* d_ws, size_t ws_size,
                              hipStream_t stream) {
    const float* e     = (const float*)d_in[0];
    const float* fc1_w = (const float*)d_in[1];
    const float* fc1_b = (const float*)d_in[2];
    const float* fc2_w = (const float*)d_in[3];
    const float* fc2_b = (const float*)d_in[4];
    const float* w_ih0 = (const float*)d_in[5];
    const float* w_hh0 = (const float*)d_in[6];
    const float* b_ih0 = (const float*)d_in[7];
    const float* b_hh0 = (const float*)d_in[8];
    const float* w_ih1 = (const float*)d_in[9];
    const float* w_hh1 = (const float*)d_in[10];
    const float* b_ih1 = (const float*)d_in[11];
    const float* b_hh1 = (const float*)d_in[12];
    const float* wv    = (const float*)d_in[13];
    const float* bv    = (const float*)d_in[14];
    const float* wt1   = (const float*)d_in[15];
    const float* bt1   = (const float*)d_in[16];
    const float* wt2   = (const float*)d_in[17];
    const float* bt2   = (const float*)d_in[18];
    const float* wm    = (const float*)d_in[19];
    const float* bm    = (const float*)d_in[20];

    char* ws = (char*)d_ws;
    size_t off = 0;
    float* h0ws = (float*)(ws + off); off += (size_t)B_TOT * 1024 * 4;
    float* gi0  = (float*)(ws + off); off += (size_t)B_TOT * GATES * 4;
    float* hid  = (float*)(ws + off); off += (size_t)B_TOT * HID * 4;
    u16* whh0b  = (u16*)(ws + off);   off += (size_t)GATES * HID * 2;
    u16* wih1b  = (u16*)(ws + off);   off += (size_t)GATES * HID * 2;
    u16* whh1b  = (u16*)(ws + off);   off += (size_t)GATES * HID * 2;
    u16* wvb    = (u16*)(ws + off);   off += (size_t)NUMF * HID * 2;
    u16* wt1b   = (u16*)(ws + off);   off += (size_t)64 * HID * 2;
    u16* wmb    = (u16*)(ws + off);   off += (size_t)NUMF * HID * 2;

    // bf16 weight conversion (one small kernel per matrix; all on `stream`)
    auto cvt = [&](const float* s, u16* d, int n) {
        k_cvt_bf16<<<(n + 255) / 256, 256, 0, stream>>>(s, d, n);
    };
    cvt(w_hh0, whh0b, GATES * HID);
    cvt(w_ih1, wih1b, GATES * HID);
    cvt(w_hh1, whh1b, GATES * HID);
    cvt(wv,    wvb,   NUMF * HID);
    cvt(wt1,   wt1b,  64 * HID);
    cvt(wm,    wmb,   NUMF * HID);

    // fc init-state chain + loop-invariant gi0
    k_fc1<<<(B_TOT * HID) / 256, 256, 0, stream>>>(e, fc1_w, fc1_b, hid);
    k_fc2<<<(B_TOT * 1024) / 256, 256, 0, stream>>>(hid, fc2_w, fc2_b, h0ws);
    k_gi0<<<(B_TOT * GATES) / 256, 256, 0, stream>>>(e, w_ih0, b_ih0, gi0);

    float* out_tn = (float*)d_out;
    float* out_u  = out_tn + (size_t)B_TOT * T_STEPS * NUMF;
    float* out_m  = out_u + (size_t)B_TOT * T_STEPS;
    int*   out_seq = (int*)(out_m + (size_t)B_TOT * T_STEPS * NUMF);

    const size_t smem = (size_t)(2 * MT * SF * 4)      // f32 states
                      + (size_t)(4 * MT * SB * 2)      // bf16 mirrors/staging
                      + (size_t)(2 * MT * NUMF * 4)    // tn, m staging
                      + (size_t)(MT * 64 * 4)          // t1 staging
                      + (size_t)(3 * MT * 4);          // act/len/stop
    (void)hipFuncSetAttribute((const void*)td_scan,
                              hipFuncAttributeMaxDynamicSharedMemorySize, (int)smem);

    td_scan<<<B_TOT / MT, 256, smem, stream>>>(
        gi0, h0ws, whh0b, wih1b, whh1b, wvb, wt1b, wmb,
        b_hh0, b_ih1, b_hh1, bv, bt1, wt2, bt2, bm,
        out_tn, out_u, out_m, out_seq);
}
